// SBA_84335977824635
// MI455X (gfx1250) — compile-verified, matrix operand order checked by reading, not run on hardware
//
#include <hip/hip_runtime.h>
#include <hip/hip_bf16.h>
#include <math.h>

// ---------------------------------------------------------------------------
// Types for CDNA5 WMMA (wave32): A/B = 16 x bf16 per lane, C/D = 8 x f32.
// ---------------------------------------------------------------------------
typedef __attribute__((ext_vector_type(8)))  float        v8f;
typedef __attribute__((ext_vector_type(16))) __bf16       v16bf;
typedef __attribute__((ext_vector_type(4)))  __bf16       v4bf;
typedef __attribute__((ext_vector_type(4)))  float        v4f;
typedef __attribute__((ext_vector_type(4)))  unsigned int u32x4;

union FragBF { u32x4 u[2]; v16bf v; };

// Load a 16x32 bf16 fragment (A layout; also valid for B loaded from a
// row-major matrix whose rows are the WMMA N index and whose columns are K).
// 16-bit layout: lane = (m | half<<4); per lane, K runs are
// {half*8 .. half*8+7} and {16+half*8 .. 16+half*8+7} -> two b128 loads.
__device__ __forceinline__ v16bf load_frag16x32(const __bf16* __restrict__ base,
                                                int ld, int row0, int k0, int lane) {
  const int m = lane & 15, h = lane >> 4;
  const __bf16* p = base + (size_t)(row0 + m) * (size_t)ld + (size_t)(k0 + h * 8);
  FragBF f;
  f.u[0] = *(const u32x4*)(p);
  f.u[1] = *(const u32x4*)(p + 16);
  return f.v;
}

__device__ __forceinline__ v8f wmma_bf16(v16bf a, v16bf b, v8f c) {
  // (neg_a, A, neg_b, B, c_mod, C, reuse_a, reuse_b)
  return __builtin_amdgcn_wmma_f32_16x16x32_bf16(false, a, false, b, (short)0, c,
                                                 false, false);
}

// ---------------------------------------------------------------------------
// Stage 0: f32 -> bf16 conversion, 4 elements/thread (b128 in, b64 out).
// ---------------------------------------------------------------------------
__global__ void __launch_bounds__(256)
cvt_f32_to_bf16(const float* __restrict__ src, __bf16* __restrict__ dst, int n4) {
  int i = blockIdx.x * 256 + threadIdx.x;
  if (i < n4) {
    v4f x = *(const v4f*)(src + 4 * (size_t)i);
    v4bf y;
    y[0] = (__bf16)x[0]; y[1] = (__bf16)x[1];
    y[2] = (__bf16)x[2]; y[3] = (__bf16)x[3];
    *(v4bf*)(dst + 4 * (size_t)i) = y;
  }
}

// ---------------------------------------------------------------------------
// GEMM core: one wave computes a 64x64 tile (4x4 WMMA tiles).
// 16 WMMAs per 16 b128 loads per k-step -> ~32 flops/byte from L2.
// Prefetch is UNCONDITIONAL and issued before the fragment loads: speculative
// prefetch past the end of the matrix is dropped by hardware (no fault, no
// counter), and keeping it branch-free avoids WMMA->VALU hazard v_nops that a
// conditional block inside the loop would incur.
// ---------------------------------------------------------------------------
__device__ __forceinline__ void gemm_wave_64x64(const __bf16* __restrict__ X,
                                                const __bf16* __restrict__ W,
                                                int m0, int n0, int lane,
                                                v8f acc[4][4]) {
  for (int k0 = 0; k0 < 2048; k0 += 32) {
    __builtin_prefetch(X + (size_t)(m0 + (lane & 15)) * 2048 + k0 + 32, 0, 3);
    __builtin_prefetch(W + (size_t)(n0 + (lane & 15)) * 2048 + k0 + 32, 0, 3);
    v16bf a[4], b[4];
#pragma unroll
    for (int t = 0; t < 4; ++t) {
      a[t] = load_frag16x32(X, 2048, m0 + 16 * t, k0, lane);
      b[t] = load_frag16x32(W, 2048, n0 + 16 * t, k0, lane);
    }
#pragma unroll
    for (int mt = 0; mt < 4; ++mt)
#pragma unroll
      for (int nt = 0; nt < 4; ++nt)
        acc[mt][nt] = wmma_bf16(a[mt], b[nt], acc[mt][nt]);
  }
}

// ---------------------------------------------------------------------------
// Stage 1: fused Q/K/V projection. C = X (2048x2048) @ W^T.
// blockIdx.z selects W in {Wq,Wk,Wv}.  Block = 4 waves -> 256x64 tile.
// Q,K stored [b,h,l,d]; V stored transposed [b,h,d,l] (for att@V B-frags).
// ---------------------------------------------------------------------------
__global__ void __launch_bounds__(128)
proj_gemm_qkv(const __bf16* __restrict__ X,
              const __bf16* __restrict__ Wq, const __bf16* __restrict__ Wk,
              const __bf16* __restrict__ Wv,
              __bf16* __restrict__ Qo, __bf16* __restrict__ Ko,
              __bf16* __restrict__ Vo) {
  const int lane = threadIdx.x & 31;
  const int wave = threadIdx.x >> 5;
  const int m0 = blockIdx.x * 256 + wave * 64;
  const int n0 = blockIdx.y * 64;
  const int which = blockIdx.z;
  const __bf16* W = (which == 0) ? Wq : (which == 1) ? Wk : Wv;
  __bf16* out     = (which == 0) ? Qo : (which == 1) ? Ko : Vo;
  const int vmode = (which == 2);

  v8f acc[4][4] = {};
  gemm_wave_64x64(X, W, m0, n0, lane, acc);

  const int half = lane >> 4, lp = lane & 15;
#pragma unroll
  for (int mt = 0; mt < 4; ++mt)
#pragma unroll
    for (int nt = 0; nt < 4; ++nt)
#pragma unroll
      for (int r = 0; r < 8; ++r) {
        int mrow = m0 + mt * 16 + r + 8 * half;   // global row  (b*1024 + i)
        int ncol = n0 + nt * 16 + lp;             // global col  (h*64 + d)
        int bb = mrow >> 10, ii = mrow & 1023, hh = ncol >> 6, dd = ncol & 63;
        size_t addr = (vmode == 0)
            ? ((size_t)((bb * 32 + hh) * 1024 + ii)) * 64 + (size_t)dd
            : ((size_t)((bb * 32 + hh) * 64 + dd)) * 1024 + (size_t)ii;
        out[addr] = (__bf16)acc[mt][nt][r];
      }
}

// ---------------------------------------------------------------------------
// Stage 3: out = O (bf16 2048x2048) @ Wo^T, f32 output row-major.
// ---------------------------------------------------------------------------
__global__ void __launch_bounds__(128)
proj_gemm_out(const __bf16* __restrict__ X, const __bf16* __restrict__ W,
              float* __restrict__ out) {
  const int lane = threadIdx.x & 31;
  const int wave = threadIdx.x >> 5;
  const int m0 = blockIdx.x * 256 + wave * 64;
  const int n0 = blockIdx.y * 64;

  v8f acc[4][4] = {};
  gemm_wave_64x64(X, W, m0, n0, lane, acc);

  const int half = lane >> 4, lp = lane & 15;
#pragma unroll
  for (int mt = 0; mt < 4; ++mt)
#pragma unroll
    for (int nt = 0; nt < 4; ++nt)
#pragma unroll
      for (int r = 0; r < 8; ++r) {
        int mrow = m0 + mt * 16 + r + 8 * half;
        int ncol = n0 + nt * 16 + lp;
        out[(size_t)mrow * 2048 + ncol] = acc[mt][nt][r];
      }
}

// ---------------------------------------------------------------------------
// Stage 2: fused stick-breaking attention.
// One wave (32 threads) per 16-query-row tile per (b,h). Key blocks of 32
// processed in DESCENDING j order (direction of the reverse exclusive cumsum).
//   z   = (Q K^T)/8                            (WMMA, bf16 in / f32 acc)
//   sp  = softplus(z);  log_sigmoid(z) = z - sp
//   lom = -sp for j<i else 0
//   s   = carry + in-block suffix-exclusive sum (cross-lane shfl scan)
//   att = exp(z - sp + s) for j<i else 0       (single v_exp_f32 per element)
//   O  += att @ V                              (WMMA via LDS re-layout)
// Q,K: [b,h,1024,64] bf16.  Vt: [b,h,64,1024] bf16.  O: [b*1024, 2048] bf16.
// ---------------------------------------------------------------------------
__global__ void __launch_bounds__(32)
sba_attn(const __bf16* __restrict__ Q, const __bf16* __restrict__ K,
         const __bf16* __restrict__ Vt, __bf16* __restrict__ O) {
  __shared__ __bf16 satt[16 * 32];

  const int lane = threadIdx.x;
  const int i0 = blockIdx.x * 16;
  const int h  = blockIdx.y;
  const int b  = blockIdx.z;
  const size_t bh = (size_t)(b * 32 + h);
  const __bf16* Qbh = Q  + bh * (size_t)(1024 * 64);
  const __bf16* Kbh = K  + bh * (size_t)(1024 * 64);
  const __bf16* Vbh = Vt + bh * (size_t)(64 * 1024);

  const int half = lane >> 4, lp = lane & 15;
  const v16bf aq0 = load_frag16x32(Qbh, 64, i0, 0,  lane);
  const v16bf aq1 = load_frag16x32(Qbh, 64, i0, 32, lane);

  v8f oacc[4] = {};
  float carry[8] = {};            // per owned row: sum of lom over keys already seen
  float tot_hi[8];
  const float scale = 0.125f;     // 1/sqrt(64)
  const int irow_base = i0 + 8 * half;

  for (int j0 = (i0 >> 5) << 5; j0 >= 0; j0 -= 32) {
#pragma unroll
    for (int nh = 1; nh >= 0; --nh) {          // high j half first (reverse scan)
      const int jn0 = j0 + nh * 16;
      v16bf bk0 = load_frag16x32(Kbh, 64, jn0, 0,  lane);
      v16bf bk1 = load_frag16x32(Kbh, 64, jn0, 32, lane);
      v8f zz = {};
      zz = wmma_bf16(aq0, bk0, zz);
      zz = wmma_bf16(aq1, bk1, zz);
#pragma unroll
      for (int r = 0; r < 8; ++r) {
        const int i = irow_base + r;
        const int j = jn0 + lp;
        const float z = zz[r] * scale;
        const bool valid = (j < i);
        // softplus via hardware transcendentals (arg of log in [1,2])
        const float sp  = fmaxf(z, 0.f) + __logf(1.f + __expf(-fabsf(z)));
        const float lom = valid ? -sp : 0.f;
        // suffix-inclusive scan over the 16-lane N segment (keys high->low)
        float incl = lom;
#pragma unroll
        for (int d = 1; d < 16; d <<= 1) {
          float y = __shfl_down(incl, d, 16);
          if (lp + d >= 16) y = 0.f;
          incl += y;
        }
        const float excl = incl - lom;
        const float tot  = __shfl(incl, half * 16, 32);  // segment lane 0 = full sum
        float s;
        if (nh == 1) { s = carry[r] + excl; tot_hi[r] = tot; }
        else         { s = carry[r] + tot_hi[r] + excl; carry[r] += tot_hi[r] + tot; }
        // att = sigmoid(z)*exp(s) = exp(z - sp + s): one exp, no divide
        const float att = valid ? __expf(z - sp + s) : 0.f;
        satt[(8 * half + r) * 32 + nh * 16 + lp] = (__bf16)att;
      }
    }
    __syncthreads();
    // Re-layout att (D format -> A format) through LDS, then O += att @ V.
    const v16bf attA = load_frag16x32(satt, 32, 0, 0, lane);
#pragma unroll
    for (int c = 0; c < 4; ++c) {
      v16bf bv = load_frag16x32(Vbh, 1024, c * 16, j0, lane);  // Vt: contiguous in j
      oacc[c] = wmma_bf16(attA, bv, oacc[c]);
    }
    __syncthreads();
  }

  // Store O tile as [b*1024 + i, h*64 + d] bf16 (input to the final GEMM).
  __bf16* Op = O + ((size_t)(b * 1024 + i0)) * 2048 + (size_t)(h * 64);
#pragma unroll
  for (int c = 0; c < 4; ++c)
#pragma unroll
    for (int r = 0; r < 8; ++r)
      Op[(size_t)(8 * half + r) * 2048 + c * 16 + lp] = (__bf16)oacc[c][r];
}

// ---------------------------------------------------------------------------
// Launcher. Workspace layout (bf16, 4M elements = 8MB each): X, Wq, Wk, Wv,
// Wo, Q[b,h,l,d], K[b,h,l,d], Vt[b,h,d,l], O[b*l, h*d]. Total 72MB.
// ---------------------------------------------------------------------------
extern "C" void kernel_launch(void* const* d_in, const int* in_sizes, int n_in,
                              void* d_out, int out_size, void* d_ws, size_t ws_size,
                              hipStream_t stream) {
  const float* hs = (const float*)d_in[0];
  const float* Wq = (const float*)d_in[1];
  const float* Wk = (const float*)d_in[2];
  const float* Wv = (const float*)d_in[3];
  const float* Wo = (const float*)d_in[4];
  float* out = (float*)d_out;

  const size_t NE = 2048ull * 2048ull;
  __bf16* Xb  = (__bf16*)d_ws;
  __bf16* Wqb = Xb  + NE;
  __bf16* Wkb = Wqb + NE;
  __bf16* Wvb = Wkb + NE;
  __bf16* Wob = Wvb + NE;
  __bf16* Qb  = Wob + NE;
  __bf16* Kb  = Qb  + NE;
  __bf16* Vtb = Kb  + NE;
  __bf16* Ob  = Vtb + NE;

  const int n4 = (int)(NE / 4);
  dim3 cg((unsigned)((n4 + 255) / 256)), cb(256);
  cvt_f32_to_bf16<<<cg, cb, 0, stream>>>(hs, Xb,  n4);
  cvt_f32_to_bf16<<<cg, cb, 0, stream>>>(Wq, Wqb, n4);
  cvt_f32_to_bf16<<<cg, cb, 0, stream>>>(Wk, Wkb, n4);
  cvt_f32_to_bf16<<<cg, cb, 0, stream>>>(Wv, Wvb, n4);
  cvt_f32_to_bf16<<<cg, cb, 0, stream>>>(Wo, Wob, n4);

  dim3 gq(2048 / 256, 2048 / 64, 3), gb(128);
  proj_gemm_qkv<<<gq, gb, 0, stream>>>(Xb, Wqb, Wkb, Wvb, Qb, Kb, Vtb);

  dim3 ag(64, 32, 2);
  sba_attn<<<ag, dim3(32), 0, stream>>>(Qb, Kb, Vtb, Ob);

  dim3 go(2048 / 256, 2048 / 64), gob(128);
  proj_gemm_out<<<go, gob, 0, stream>>>(Ob, Wob, out);
}
